// SemanticAwareAttention_68470368633523
// MI455X (gfx1250) — matrix operand order repaired and checked by reading, not compiled, not run on hardware
//
#include <hip/hip_runtime.h>
#include <math.h>

// ---------------------------------------------------------------------------
// SemanticAwareAttention fused kernel for MI455X (gfx1250, wave32, WMMA+TDM).
//   qp = q @ Wq^T + bq                      (131072 x 512) -- dominant GEMM
//   kp = k @ Wk^T + bk ; vp = v @ Wv^T + bv (100 x 512, tiny -> prep kernels)
//   a1 = softmax(qp kp^T / sqrt(512))
//   a2 = softmax(a1 * softmax(coarse_pred))
//   out = a2 @ vp
// bf16 WMMA (v_wmma_f32_16x16x32_bf16) with f32 accumulation keeps the main
// GEMM near the 23.3 TB/s memory roofline (~590 MB min traffic ~= 25 us).
// bf16 staging (kp / Wq slices / vpT) rides the Tensor Data Mover:
// double-buffered tensor_load_to_lds overlapped with WMMA / softmax compute,
// fenced with s_wait_tensorcnt + workgroup barriers.
// ---------------------------------------------------------------------------

typedef __bf16 bf16_t;
typedef __attribute__((ext_vector_type(16))) __bf16 v16bf;
typedef __attribute__((ext_vector_type(8)))  __bf16 v8bf;
typedef __attribute__((ext_vector_type(4)))  __bf16 v4bf;
typedef __attribute__((ext_vector_type(8)))  float  v8f;
typedef __attribute__((ext_vector_type(4)))  float  v4f;
typedef __attribute__((ext_vector_type(4)))  unsigned int u32x4;
typedef __attribute__((ext_vector_type(8)))  int          i32x8;
typedef __attribute__((ext_vector_type(4)))  int          i32x4;

#define EMBED   512
#define NCLS    100
#define NCLSP   112   // classes padded to 7*16 for WMMA N-tiles
#define KPAD    128   // attn K padded to 4*32 for WMMA K-steps
#define ROWS_PB 128   // rows per block (8 waves x 16)
#define NEGBIG  (-3.0e38f)

// LDS layout (bytes), total 319488 B <= 320 KB WGP budget:
//   s_q    @      0 : 131072  q tile bf16 128x512   | phase 3: vpT 512x128
//   s_kp   @ 131072 : 114688  kp bf16 112x512
//   s_wq0  @ 245760 :  32768  Wq slice ping (32x512) | phase 2+: attn scratch
//   s_wq1  @ 278528 :  32768  Wq slice pong (32x512)
//   s_qtr  @ 311296 :   8192  per-wave qp relayout scratch 8 x 16x32
#define LDS_OFF_KP   131072u
#define LDS_OFF_WQ0  245760u
#define LDS_OFF_WQ1  278528u
#define LDS_OFF_QTR  311296u
#define LDS_BYTES    319488u

#if defined(__gfx1250__) && __has_builtin(__builtin_amdgcn_tensor_load_to_lds)
#define USE_TDM 1
#else
#define USE_TDM 0
#endif

static __device__ __forceinline__ v16bf cat8(v8bf a, v8bf b) {
  return __builtin_shufflevector(a, b, 0,1,2,3,4,5,6,7,8,9,10,11,12,13,14,15);
}
static __device__ __forceinline__ v8f wmma_bf16(v16bf a, v16bf b, v8f c) {
  // D = A(16x32) * B(32x16) + C, f32 accumulate
  return __builtin_amdgcn_wmma_f32_16x16x32_bf16(false, a, false, b,
                                                 (short)0, c, false, false);
}
// A-matrix fragment (16x32 bf16): lane = M (lane&15); lanes<16 hold K {0..7,16..23},
// lanes>=16 hold K {8..15,24..31} (per CDNA5 ISA 7.12.2).  base points at row 0, K 0.
static __device__ __forceinline__ v16bf load_A(const bf16_t* base, int strideH,
                                               int ln, int hi) {
  const bf16_t* p = base + ln * strideH;
  v8bf lo = *(const v8bf*)(p + (hi ? 8  : 0));
  v8bf hh = *(const v8bf*)(p + (hi ? 24 : 16));
  return cat8(lo, hh);
}
// B-matrix fragment (32x16 bf16): lane = N (lane&15); lanes<16 hold K {0..7,16..23},
// lanes>=16 hold K {8..15,24..31} (same order as A). base points at col-row 0, K 0.
static __device__ __forceinline__ v16bf load_B(const bf16_t* base, int strideH,
                                               int ln, int hi) {
  const bf16_t* p = base + ln * strideH;
  return cat8(*(const v8bf*)(p + (hi ? 8 : 0)), *(const v8bf*)(p + (hi ? 24 : 16)));
}

#if USE_TDM
// Issue a 2D Tensor-Data-Mover load (bf16 elements) into LDS.
// D# per CDNA5 ISA 8.3/8.4: group0 = {count/type | lds_addr | global_addr},
// group1 = {data_size | tensor dims | tile dims | dim0 stride}; 2D tensors
// leave groups 2/3 (and the trailing extension group of the 6-arg builtin)
// null.  Completion tracked on TENSORcnt.
static __device__ __forceinline__ void tdm_load_2d(unsigned lds_off,
                                                   const bf16_t* gsrc,
                                                   unsigned tensor_d0,
                                                   unsigned tensor_d1,
                                                   unsigned tile_d0,
                                                   unsigned tile_d1,
                                                   unsigned stride0) {
  unsigned long long ga = (unsigned long long)(const void*)gsrc;
  u32x4 g0;
  g0[0] = 1u;                                         // count=1, user D#
  g0[1] = lds_off;                                    // lds_addr (bytes)
  g0[2] = (unsigned)(ga & 0xffffffffu);               // global_addr[31:0]
  g0[3] = (unsigned)((ga >> 32) & 0x1ffffffu)         // global_addr[56:32]
          | 0x80000000u;                              // type = 2 ("image")
  i32x8 g1;
  g1[0] = (int)(1u << 16);                            // data_size = 2 bytes
  g1[1] = (int)(tensor_d0 << 16);                     // tensor_dim0[15:0]
  g1[2] = (int)((tensor_d0 >> 16) | (tensor_d1 << 16));
  g1[3] = (int)((tensor_d1 >> 16) | (tile_d0 << 16)); // tile_dim0
  g1[4] = (int)tile_d1;                               // tile_dim1 (tile_dim2=0)
  g1[5] = (int)stride0;                               // tensor_dim0_stride[31:0]
  g1[6] = 0;                                          // stride0[47:32]=0, stride1
  g1[7] = 0;
  i32x4 gz4 = {0, 0, 0, 0};
  i32x8 gz8 = {0, 0, 0, 0, 0, 0, 0, 0};
  // clang-23 / therock-10.0 signature: (u32x4, i32x8, i32x4, i32x4, i32x8, i32)
  __builtin_amdgcn_tensor_load_to_lds(g0, g1, gz4, gz4, gz8, 0);
}
#endif

// ---------------------------- prep kernels ---------------------------------

__global__ __launch_bounds__(256) void prep_wq_bf16(const float* __restrict__ Wq,
                                                    bf16_t* __restrict__ dst) {
  int i = (blockIdx.x * 256 + threadIdx.x) * 4;   // 512*512 elems, 256 blocks
  v4f f = *(const v4f*)(Wq + i);
  v4bf b; b[0]=(bf16_t)f[0]; b[1]=(bf16_t)f[1]; b[2]=(bf16_t)f[2]; b[3]=(bf16_t)f[3];
  *(v4bf*)(dst + i) = b;
}

__global__ __launch_bounds__(256) void prep_kproj(const float* __restrict__ k,
                                                  const float* __restrict__ Wk,
                                                  const float* __restrict__ bk,
                                                  bf16_t* __restrict__ kp) {
  int idx = blockIdx.x * 256 + threadIdx.x;       // 112*512 outputs, 224 blocks
  int c = idx >> 9, d = idx & 511;
  float acc = 0.f;
  if (c < NCLS) {
    const float* kr = k + c * EMBED;
    const float* wr = Wk + d * EMBED;
    for (int j = 0; j < EMBED; j += 4) {
      v4f a = *(const v4f*)(kr + j);
      v4f w = *(const v4f*)(wr + j);
      acc += a[0]*w[0] + a[1]*w[1] + a[2]*w[2] + a[3]*w[3];
    }
    acc += bk[d];
  }
  kp[idx] = (bf16_t)acc;   // kp_bf16[c][d], rows >= 100 are zero
}

__global__ __launch_bounds__(256) void prep_vproj(const float* __restrict__ v,
                                                  const float* __restrict__ Wv,
                                                  const float* __restrict__ bv,
                                                  bf16_t* __restrict__ vpt) {
  int idx = blockIdx.x * 256 + threadIdx.x;       // 512*128 outputs, 256 blocks
  int d = idx >> 7, c = idx & 127;
  float acc = 0.f;
  if (c < NCLS) {
    const float* vr = v + c * EMBED;
    const float* wr = Wv + d * EMBED;
    for (int j = 0; j < EMBED; j += 4) {
      v4f a = *(const v4f*)(vr + j);
      v4f w = *(const v4f*)(wr + j);
      acc += a[0]*w[0] + a[1]*w[1] + a[2]*w[2] + a[3]*w[3];
    }
    acc += bv[d];
  }
  vpt[idx] = (bf16_t)acc;  // vpT_bf16[d][c] (column c = class, padded zeros)
}

// ----------------------------- main kernel ---------------------------------

__global__ __launch_bounds__(256) void saa_main(const float*  __restrict__ q,
                                                const float*  __restrict__ cpred,
                                                const bf16_t* __restrict__ wq_bf,
                                                const float*  __restrict__ bq,
                                                const bf16_t* __restrict__ kp_bf,
                                                const bf16_t* __restrict__ vpt_bf,
                                                float*        __restrict__ out) {
  extern __shared__ bf16_t smem[];
  bf16_t* s_q   = smem;                              // 128x512 bf16
  bf16_t* s_kp  = smem + (LDS_OFF_KP  / 2);          // 112x512 bf16
  bf16_t* s_wq0 = smem + (LDS_OFF_WQ0 / 2);          // 32x512 bf16 (ping)
  bf16_t* s_wq1 = smem + (LDS_OFF_WQ1 / 2);          // 32x512 bf16 (pong)
  bf16_t* s_qtr = smem + (LDS_OFF_QTR / 2);          // 8 x 16x32 bf16
  bf16_t* s_vpt = s_q;                               // phase-3 union: 512x128
  bf16_t* s_atr = s_wq0;                             // phase-2+ union: 8 x 16x128

  const int tid  = threadIdx.x;
  const int lane = tid & 31;
  const int wave = tid >> 5;
  const int ln   = lane & 15;
  const int hi   = lane >> 4;
  const int row0 = blockIdx.x * ROWS_PB;
  const int myrow0 = row0 + wave * 16;
  bf16_t* my_qtr = s_qtr + wave * 16 * 32;
  bf16_t* my_atr = s_atr + wave * 16 * KPAD;

  // ---- prologue: TDM kp + Wq slice 0 while all threads convert q ----------
#if USE_TDM
  if (tid == 0) {
    tdm_load_2d(LDS_OFF_KP,  kp_bf, EMBED, NCLSP, EMBED, NCLSP, EMBED);
    tdm_load_2d(LDS_OFF_WQ0, wq_bf, EMBED, EMBED, EMBED, 32,    EMBED);
  }
#else
  for (int i = 0; i < 28; ++i) {                     // 112*512/8/256
    int idx = (i * 256 + tid) * 8;
    *(v8bf*)(s_kp + idx) = *(const v8bf*)(kp_bf + idx);
  }
  for (int i = 0; i < 8; ++i) {                      // 32*512/8/256
    int idx = (i * 256 + tid) * 8;
    *(v8bf*)(s_wq0 + idx) = *(const v8bf*)(wq_bf + idx);
  }
#endif
  {
    const float* qsrc = q + (size_t)row0 * EMBED;    // q f32 -> bf16, 128x512
    for (int i = 0; i < 64; ++i) {
      int idx = i * 1024 + tid * 4;
      v4f f = *(const v4f*)(qsrc + idx);
      v4bf b; b[0]=(bf16_t)f[0]; b[1]=(bf16_t)f[1]; b[2]=(bf16_t)f[2]; b[3]=(bf16_t)f[3];
      *(v4bf*)(s_q + idx) = b;
    }
  }
#if USE_TDM
  if (tid == 0) __builtin_amdgcn_s_wait_tensorcnt(0);
#endif
  __syncthreads();

  const float scale = 0.044194173824159216f;         // 1/sqrt(512)
  v8f la[7];                                         // logits, 16 rows x 112
  #pragma unroll
  for (int c = 0; c < 7; ++c) la[c] = (v8f){};

  // ---- fused projection + logits; Wq slices double-buffered via TDM -------
  for (int n2 = 0; n2 < 16; ++n2) {
    bf16_t* wqb = (n2 & 1) ? s_wq1 : s_wq0;
    // kick off DMA of the NEXT 32-row Wq slice under this iteration's WMMAs
#if USE_TDM
    if (tid == 0 && n2 < 15)
      tdm_load_2d((n2 & 1) ? LDS_OFF_WQ0 : LDS_OFF_WQ1,
                  wq_bf + (n2 + 1) * 32 * EMBED, EMBED, EMBED, EMBED, 32, EMBED);
#else
    if (n2 < 15) {
      bf16_t* nxt = (n2 & 1) ? s_wq0 : s_wq1;
      for (int i = 0; i < 8; ++i) {
        int idx = (i * 256 + tid) * 8;
        *(v8bf*)(nxt + idx) = *(const v8bf*)(wq_bf + (n2 + 1) * 32 * EMBED + idx);
      }
    }
#endif

    v8f acc0 = (v8f){}, acc1 = (v8f){};
    const bf16_t* aBase = s_q + (wave * 16) * EMBED;
    #pragma unroll
    for (int ks = 0; ks < 16; ++ks) {                // K = 512 in steps of 32
      v16bf A  = load_A(aBase + ks * 32, EMBED, ln, hi);
      v16bf B0 = load_B(wqb + ks * 32, EMBED, ln, hi);
      v16bf B1 = load_B(wqb + 16 * EMBED + ks * 32, EMBED, ln, hi);
      acc0 = wmma_bf16(A, B0, acc0);
      acc1 = wmma_bf16(A, B1, acc1);
    }
    // bias (C-layout: col = lane&15 within tile), then C->A relayout via LDS
    float bb0 = bq[n2 * 32 + ln];
    float bb1 = bq[n2 * 32 + 16 + ln];
    #pragma unroll
    for (int r = 0; r < 8; ++r) {
      int rr = r + 8 * hi;
      my_qtr[rr * 32 + ln]      = (bf16_t)(acc0[r] + bb0);
      my_qtr[rr * 32 + 16 + ln] = (bf16_t)(acc1[r] + bb1);
    }
    asm volatile("s_wait_dscnt 0" ::: "memory");
    v16bf QA = load_A(my_qtr, 32, ln, hi);           // qp fragment, 16x32
    #pragma unroll
    for (int ct = 0; ct < 7; ++ct) {                 // logits += qp * kp^T
      v16bf B = load_B(s_kp + (ct * 16) * EMBED + n2 * 32, EMBED, ln, hi);
      la[ct] = wmma_bf16(QA, B, la[ct]);
    }
#if USE_TDM
    if (tid == 0) __builtin_amdgcn_s_wait_tensorcnt(0);  // next slice landed
#endif
    __syncthreads();                                 // slice consumed+published
  }

  // ---- start vpT DMA (s_q now free), hide it under the softmax phase ------
#if USE_TDM
  if (tid == 0)
    tdm_load_2d(0u, vpt_bf, KPAD, EMBED, KPAD, EMBED, KPAD);
#endif

  // ---- softmax( softmax(logits/sqrt(d)) * softmax(coarse) ) per row -------
  // C-layout: row = r + 8*hi; a row's 16 classes per tile live across the 16
  // lanes of a half-wave -> xor-butterfly reductions over offsets 8..1.
  #pragma unroll
  for (int r = 0; r < 8; ++r) {
    int row = myrow0 + r + 8 * hi;
    float x[7], e[7];
    float m = NEGBIG;
    #pragma unroll
    for (int ct = 0; ct < 7; ++ct) {
      x[ct] = (ct * 16 + ln) < NCLS ? la[ct][r] * scale : NEGBIG;
      m = fmaxf(m, x[ct]);
    }
    for (int off = 8; off; off >>= 1) m = fmaxf(m, __shfl_xor(m, off, 32));
    float s = 0.f;
    #pragma unroll
    for (int ct = 0; ct < 7; ++ct) {
      e[ct] = (ct * 16 + ln) < NCLS ? __expf(x[ct] - m) : 0.f; s += e[ct];
    }
    for (int off = 8; off; off >>= 1) s += __shfl_xor(s, off, 32);
    float inv = 1.f / s;

    float cx[7], ce[7];
    float cm = NEGBIG;
    #pragma unroll
    for (int ct = 0; ct < 7; ++ct) {
      int cls = ct * 16 + ln;
      cx[ct] = (cls < NCLS) ? cpred[(size_t)row * NCLS + cls] : NEGBIG;
      cm = fmaxf(cm, cx[ct]);
    }
    for (int off = 8; off; off >>= 1) cm = fmaxf(cm, __shfl_xor(cm, off, 32));
    float cs = 0.f;
    #pragma unroll
    for (int ct = 0; ct < 7; ++ct) {
      ce[ct] = (ct * 16 + ln) < NCLS ? __expf(cx[ct] - cm) : 0.f; cs += ce[ct];
    }
    for (int off = 8; off; off >>= 1) cs += __shfl_xor(cs, off, 32);
    float cinv = 1.f / cs;

    float t[7], e2[7];
    float m2 = NEGBIG;
    #pragma unroll
    for (int ct = 0; ct < 7; ++ct) {
      t[ct] = (ct * 16 + ln) < NCLS ? (e[ct] * inv) * (ce[ct] * cinv) : NEGBIG;
      m2 = fmaxf(m2, t[ct]);
    }
    for (int off = 8; off; off >>= 1) m2 = fmaxf(m2, __shfl_xor(m2, off, 32));
    float s2 = 0.f;
    #pragma unroll
    for (int ct = 0; ct < 7; ++ct) {
      e2[ct] = (ct * 16 + ln) < NCLS ? __expf(t[ct] - m2) : 0.f; s2 += e2[ct];
    }
    for (int off = 8; off; off >>= 1) s2 += __shfl_xor(s2, off, 32);
    float i2 = 1.f / s2;

    int rr = r + 8 * hi;                             // attn2 -> bf16 scratch
    #pragma unroll
    for (int ct = 0; ct < 7; ++ct)
      my_atr[rr * KPAD + ct * 16 + ln] = (bf16_t)(e2[ct] * i2);
    my_atr[rr * KPAD + NCLSP + ln] = (bf16_t)0.f;    // zero pad cols 112..127
  }

  // ---- phase 3: out = attn2(16x128) @ vp(128x512) from vpT in LDS ---------
#if USE_TDM
  if (tid == 0) __builtin_amdgcn_s_wait_tensorcnt(0);
#else
  __syncthreads();
  for (int i = 0; i < 32; ++i) {                     // 512*128/8/256
    int idx = (i * 256 + tid) * 8;
    *(v8bf*)(s_vpt + idx) = *(const v8bf*)(vpt_bf + idx);
  }
#endif
  __syncthreads();

  v16bf AT[4];
  #pragma unroll
  for (int kk = 0; kk < 4; ++kk)
    AT[kk] = load_A(my_atr + kk * 32, KPAD, ln, hi);

  float* orow = out + (size_t)myrow0 * EMBED;
  for (int nt = 0; nt < 32; ++nt) {
    v8f acc = (v8f){};
    #pragma unroll
    for (int kk = 0; kk < 4; ++kk) {
      // B[k][n] = vp[k][n] = vpT[n][k]: lane = out-col, contiguous K run
      v16bf B = load_B(s_vpt + (nt * 16) * KPAD + kk * 32, KPAD, ln, hi);
      acc = wmma_bf16(AT[kk], B, acc);
    }
    #pragma unroll
    for (int r = 0; r < 8; ++r)
      orow[(size_t)(r + 8 * hi) * EMBED + nt * 16 + ln] = acc[r];
  }
}

// ------------------------------- launcher ----------------------------------

extern "C" void kernel_launch(void* const* d_in, const int* in_sizes, int n_in,
                              void* d_out, int out_size, void* d_ws, size_t ws_size,
                              hipStream_t stream) {
  const float* q   = (const float*)d_in[0];
  const float* k   = (const float*)d_in[1];
  const float* v   = (const float*)d_in[2];
  const float* cp  = (const float*)d_in[3];
  const float* Wq  = (const float*)d_in[4];
  const float* bq  = (const float*)d_in[5];
  const float* Wk  = (const float*)d_in[6];
  const float* bk  = (const float*)d_in[7];
  const float* Wv  = (const float*)d_in[8];
  const float* bv  = (const float*)d_in[9];
  float* out = (float*)d_out;

  // workspace: Wq bf16 (512 KB) | kp bf16 112x512 (112 KB) | vpT bf16 512x128 (128 KB)
  bf16_t* wq_bf  = (bf16_t*)d_ws;
  bf16_t* kp_bf  = wq_bf + 512 * 512;
  bf16_t* vpt_bf = kp_bf + NCLSP * EMBED;

  prep_wq_bf16<<<256, 256, 0, stream>>>(Wq, wq_bf);
  prep_kproj  <<<224, 256, 0, stream>>>(k, Wk, bk, kp_bf);
  prep_vproj  <<<256, 256, 0, stream>>>(v, Wv, bv, vpt_bf);

  saa_main<<<131072 / ROWS_PB, 256, LDS_BYTES, stream>>>(q, cp, wq_bf, bq,
                                                         kp_bf, vpt_bf, out);
}
